// HarmonicOscillatorWithInteraction1D_89137751261222
// MI455X (gfx1250) — matrix operand order repaired and checked by reading, not compiled, not run on hardware
//
#include <hip/hip_runtime.h>
#include <hip/hip_bf16.h>

// ---------------------------------------------------------------------------
// HarmonicOscillatorWithInteraction1D local energy, CDNA5 (gfx1250) WMMA path.
// B=8192 walkers, N=8 particles, H=512 hidden. Compute-bound (~42 GFLOP fp16
// GEMMs vs ~75MB traffic @ 23.3 TB/s), so all H x H contractions run through
// v_wmma_f32_16x16x32_f16 with fp32 accumulation.
// ---------------------------------------------------------------------------

typedef _Float16 v16h __attribute__((ext_vector_type(16)));
typedef _Float16 v8h  __attribute__((ext_vector_type(8)));
typedef float    v8f  __attribute__((ext_vector_type(8)));

#define Bn 8192
#define Nn 8
#define Hn 512

// ----- WMMA fragment helpers (layouts per CDNA5 ISA 7.12.2) -----------------

// A-matrix 16x32 f16: lanes 0-15 hold row M=lane, K in {k0..k0+7, k0+16..k0+23};
// lanes 16-31 hold row M=lane-16, K in {k0+8..k0+15, k0+24..k0+31}.
__device__ __forceinline__ v16h load_afrag(const _Float16* __restrict__ base,
                                           int m0, int k0, int lane) {
  const int hi = lane >> 4, lo = lane & 15;
  const _Float16* p = base + (size_t)(m0 + lo) * Hn + k0 + hi * 8;
  v8h x0 = *(const v8h*)(p);
  v8h x1 = *(const v8h*)(p + 16);
  return __builtin_shufflevector(x0, x1, 0,1,2,3,4,5,6,7,8,9,10,11,12,13,14,15);
}

// B-matrix 32x16 f16 stored "N-major" (row n = contiguous K values):
// lanes 0-15 hold column N=lane, K in {k0..k0+7, k0+16..k0+23}; lanes 16-31 K in {k0+8..k0+15, k0+24..k0+31}.
__device__ __forceinline__ v16h load_bfrag(const _Float16* __restrict__ base,
                                           int n0, int k0, int lane) {
  const int hi = lane >> 4, lo = lane & 15;
  const _Float16* p = base + (size_t)(n0 + lo) * Hn + k0 + hi * 8;
  v8h x0 = *(const v8h*)(p);
  v8h x1 = *(const v8h*)(p + 16);
  return __builtin_shufflevector(x0, x1, 0,1,2,3,4,5,6,7,8,9,10,11,12,13,14,15);
}

__device__ __forceinline__ v8f wmma16(v16h a, v16h b, v8f c) {
  return __builtin_amdgcn_wmma_f32_16x16x32_f16(false, a, false, b,
                                                (short)0, c, false, false);
}

// ----- K0: weight prep: W2 -> f16 (row & transposed), W1 -> f16, q ----------
__global__ void k0_prep(const float* __restrict__ W1, const float* __restrict__ W2,
                        _Float16* __restrict__ w2h, _Float16* __restrict__ w2th,
                        _Float16* __restrict__ w1h, float* __restrict__ q) {
  int t = blockIdx.x * blockDim.x + threadIdx.x;
  if (t < Hn * Hn) {
    int j = t >> 9, k = t & (Hn - 1);
    float v = W2[t];
    w2h[t] = (_Float16)v;               // row-major  [j,k]
    w2th[(size_t)k * Hn + j] = (_Float16)v;  // transposed [k,j]
  }
  if (t < Hn) {
    float s = 0.f;
#pragma unroll
    for (int n = 0; n < Nn; ++n) {
      float w = W1[n * Hn + t];
      w1h[n * Hn + t] = (_Float16)w;
      s = fmaf(w, w, s);
    }
    q[t] = s;
  }
}

// ----- K1: forward layer 1 (K=8, scalar), h1/s1 outputs; zero lapA ----------
__global__ void k1_fwd(const float* __restrict__ x, const float* __restrict__ W1,
                       const float* __restrict__ b1,
                       float* __restrict__ h1f, _Float16* __restrict__ h1h,
                       _Float16* __restrict__ s1h, float* __restrict__ lapA) {
  int t = blockIdx.x * blockDim.x + threadIdx.x;
  if (t < Bn * Hn) {
    int b = t >> 9, j = t & (Hn - 1);
    float z = b1[j];
#pragma unroll
    for (int n = 0; n < Nn; ++n) z = fmaf(x[b * Nn + n], W1[n * Hn + j], z);
    float h = tanhf(z);
    float s = fmaf(-h, h, 1.f);
    h1f[t] = h;
    h1h[t] = (_Float16)h;
    s1h[t] = (_Float16)s;
  }
  if (t < Bn) lapA[t] = 0.f;
}

// ----- K2: Z2 = h1 @ W2 (WMMA), epilogue: v2 = s2*W3 (f16), w = W3*h2*s2 ----
__global__ __launch_bounds__(128)
void k2_gemm1(const _Float16* __restrict__ h1h, const _Float16* __restrict__ w2th,
              const float* __restrict__ b2, const float* __restrict__ W3,
              _Float16* __restrict__ v2h, float* __restrict__ wf) {
  const int lane = threadIdx.x & 31;
  const int gw = blockIdx.x * (blockDim.x >> 5) + (threadIdx.x >> 5);
  const int wm = gw >> 4;          // 256 tiles of 32 walkers
  const int wn = gw & 15;          // 16 tiles of 32 cols
  const int m0 = wm * 32, n0 = wn * 32;

  v8f acc00 = {}, acc01 = {}, acc10 = {}, acc11 = {};
  for (int k0 = 0; k0 < Hn; k0 += 32) {
    v16h a0 = load_afrag(h1h, m0, k0, lane);
    v16h a1 = load_afrag(h1h, m0 + 16, k0, lane);
    v16h bb0 = load_bfrag(w2th, n0, k0, lane);
    v16h bb1 = load_bfrag(w2th, n0 + 16, k0, lane);
    acc00 = wmma16(a0, bb0, acc00);
    acc01 = wmma16(a0, bb1, acc01);
    acc10 = wmma16(a1, bb0, acc10);
    acc11 = wmma16(a1, bb1, acc11);
  }
  const int hi = lane >> 4, lo = lane & 15;
#pragma unroll
  for (int mi = 0; mi < 2; ++mi) {
#pragma unroll
    for (int ni = 0; ni < 2; ++ni) {
      v8f acc = (mi == 0) ? (ni == 0 ? acc00 : acc01) : (ni == 0 ? acc10 : acc11);
      int k = n0 + ni * 16 + lo;
      float bk = b2[k], w3 = W3[k];
#pragma unroll
      for (int v = 0; v < 8; ++v) {
        int br = m0 + mi * 16 + v + 8 * hi;
        float z = acc[v] + bk;
        float h2 = tanhf(z);
        float s2 = fmaf(-h2, h2, 1.f);
        v2h[(size_t)br * Hn + k] = (_Float16)(s2 * w3);
        wf[(size_t)br * Hn + k] = w3 * h2 * s2;
      }
    }
  }
}

// ----- K3: R = v2 @ W2^T (WMMA): R[b,j] = sum_k v2[b,k] W2[j,k] -------------
__global__ __launch_bounds__(128)
void k3_gemm2(const _Float16* __restrict__ v2h, const _Float16* __restrict__ w2h,
              float* __restrict__ Rf) {
  const int lane = threadIdx.x & 31;
  const int gw = blockIdx.x * (blockDim.x >> 5) + (threadIdx.x >> 5);
  const int wm = gw >> 4;
  const int wn = gw & 15;
  const int m0 = wm * 32, n0 = wn * 32;

  v8f acc00 = {}, acc01 = {}, acc10 = {}, acc11 = {};
  for (int k0 = 0; k0 < Hn; k0 += 32) {
    v16h a0 = load_afrag(v2h, m0, k0, lane);
    v16h a1 = load_afrag(v2h, m0 + 16, k0, lane);
    v16h bb0 = load_bfrag(w2h, n0, k0, lane);   // row j of W2 is contiguous in k
    v16h bb1 = load_bfrag(w2h, n0 + 16, k0, lane);
    acc00 = wmma16(a0, bb0, acc00);
    acc01 = wmma16(a0, bb1, acc01);
    acc10 = wmma16(a1, bb0, acc10);
    acc11 = wmma16(a1, bb1, acc11);
  }
  const int hi = lane >> 4, lo = lane & 15;
#pragma unroll
  for (int mi = 0; mi < 2; ++mi) {
#pragma unroll
    for (int ni = 0; ni < 2; ++ni) {
      v8f acc = (mi == 0) ? (ni == 0 ? acc00 : acc01) : (ni == 0 ? acc10 : acc11);
      int j = n0 + ni * 16 + lo;
#pragma unroll
      for (int v = 0; v < 8; ++v) {
        int br = m0 + mi * 16 + v + 8 * hi;
        Rf[(size_t)br * Hn + j] = acc[v];
      }
    }
  }
}

// ----- K5: Laplacian first term. Per 16x16 tile, 8 live accumulators (n) ----
// A_n = (s1 * W1_row_n) @ W2 ; lapA[b] += sum_k w[b,k] * sum_n A_n[b,k]^2
__global__ __launch_bounds__(128)
void k5_lap(const _Float16* __restrict__ s1h, const _Float16* __restrict__ w2th,
            const _Float16* __restrict__ w1h, const float* __restrict__ wf,
            float* __restrict__ lapA) {
  __shared__ _Float16 sW1[Nn * Hn];
  {
    const uint4* src = (const uint4*)w1h;
    uint4* dst = (uint4*)sW1;
    for (int i = threadIdx.x; i < (Nn * Hn) / 8; i += blockDim.x) dst[i] = src[i];
  }
  __syncthreads();

  const int lane = threadIdx.x & 31;
  const int gw = blockIdx.x * (blockDim.x >> 5) + (threadIdx.x >> 5);
  const int wm = gw >> 5;          // 512 tiles of 16 walkers
  const int wn = gw & 31;          // 32 tiles of 16 cols
  const int m0 = wm * 16, n0 = wn * 16;
  const int hi = lane >> 4;

  v8f acc[Nn] = {};
  for (int k0 = 0; k0 < Hn; k0 += 32) {
    v16h s = load_afrag(s1h, m0, k0, lane);
    v16h bb = load_bfrag(w2th, n0, k0, lane);
#pragma unroll
    for (int n = 0; n < Nn; ++n) {
      const _Float16* p = sW1 + n * Hn + k0 + hi * 8;
      v8h w0 = *(const v8h*)(p);
      v8h w1x = *(const v8h*)(p + 16);
      v16h wv = __builtin_shufflevector(w0, w1x,
                    0,1,2,3,4,5,6,7,8,9,10,11,12,13,14,15);
      v16h u = s * wv;                 // v_pk_mul_f16: U_n = s1 * W1[n,:]
      acc[n] = wmma16(u, bb, acc[n]);
    }
  }

  v8f sq = {};
#pragma unroll
  for (int n = 0; n < Nn; ++n)
#pragma unroll
    for (int v = 0; v < 8; ++v) sq[v] = fmaf(acc[n][v], acc[n][v], sq[v]);

  const int lo = lane & 15;
#pragma unroll
  for (int v = 0; v < 8; ++v) {
    int br = m0 + v + 8 * hi;
    int k = n0 + lo;
    float val = wf[(size_t)br * Hn + k] * sq[v];
    val += __shfl_xor(val, 1, 32);
    val += __shfl_xor(val, 2, 32);
    val += __shfl_xor(val, 4, 32);
    val += __shfl_xor(val, 8, 32);
    if (lo == 0) atomicAdd(&lapA[br], val);
  }
}

// ----- K4: per-walker finalize: g, term2, pot, vint, assemble output --------
__global__ __launch_bounds__(256)
void k4_final(const float* __restrict__ x, const float* __restrict__ W1,
              const float* __restrict__ q, const float* __restrict__ h1f,
              const float* __restrict__ Rf, const float* __restrict__ lapA,
              float* __restrict__ out) {
  const int lane = threadIdx.x & 31;
  const int b = blockIdx.x * (blockDim.x >> 5) + (threadIdx.x >> 5);

  float accn[Nn] = {};
  float accT2 = 0.f;
  for (int it = 0; it < Hn / 32; ++it) {
    int j = lane + it * 32;
    float r = Rf[(size_t)b * Hn + j];
    float h = h1f[(size_t)b * Hn + j];
    float s = fmaf(-h, h, 1.f);
    float v1 = s * r;
#pragma unroll
    for (int n = 0; n < Nn; ++n) accn[n] = fmaf(v1, W1[n * Hn + j], accn[n]);
    accT2 = fmaf(h * s * q[j], r, accT2);
  }
  float g2 = 0.f;
#pragma unroll
  for (int n = 0; n < Nn; ++n) {
    float a = accn[n];
    a += __shfl_xor(a, 1, 32);
    a += __shfl_xor(a, 2, 32);
    a += __shfl_xor(a, 4, 32);
    a += __shfl_xor(a, 8, 32);
    a += __shfl_xor(a, 16, 32);
    g2 = fmaf(a, a, g2);
  }
  accT2 += __shfl_xor(accT2, 1, 32);
  accT2 += __shfl_xor(accT2, 2, 32);
  accT2 += __shfl_xor(accT2, 4, 32);
  accT2 += __shfl_xor(accT2, 8, 32);
  accT2 += __shfl_xor(accT2, 16, 32);

  if (lane == 0) {
    float term1 = -2.f * lapA[b];
    float term2 = -2.f * accT2;
    float ek = -0.5f * (term1 + term2 + g2);

    float xi[Nn];
    float sumx2 = 0.f;
#pragma unroll
    for (int i = 0; i < Nn; ++i) {
      xi[i] = x[b * Nn + i];
      sumx2 = fmaf(xi[i], xi[i], sumx2);
    }
    float sg = 0.f;
#pragma unroll
    for (int i = 0; i < Nn; ++i)
#pragma unroll
      for (int j2 = i + 1; j2 < Nn; ++j2) {
        float d = xi[j2] - xi[i];
        sg += expf(-d * d * 2.0f);   // 1/(2*sigma0^2) = 2
      }
    const float GC = 1.5957691216057308f;  // V0/(sqrt(2pi)*sigma0)
    out[b] = ek + 0.5f * sumx2 + GC * sg;
  }
}

extern "C" void kernel_launch(void* const* d_in, const int* in_sizes, int n_in,
                              void* d_out, int out_size, void* d_ws, size_t ws_size,
                              hipStream_t stream) {
  const float* x  = (const float*)d_in[0];
  const float* W1 = (const float*)d_in[1];
  const float* b1 = (const float*)d_in[2];
  const float* W2 = (const float*)d_in[3];
  const float* b2 = (const float*)d_in[4];
  const float* W3 = (const float*)d_in[5];
  // d_in[6] = b3: constant shift of logpsi, cancels in all derivatives.

  size_t off = 0;
  auto alloc = [&](size_t bytes) -> void* {
    off = (off + 255) & ~(size_t)255;
    void* r = (char*)d_ws + off;
    off += bytes;
    return r;
  };
  _Float16* w2h  = (_Float16*)alloc((size_t)Hn * Hn * 2);
  _Float16* w2th = (_Float16*)alloc((size_t)Hn * Hn * 2);
  _Float16* w1h  = (_Float16*)alloc((size_t)Nn * Hn * 2);
  float*    q    = (float*)   alloc((size_t)Hn * 4);
  float*    h1f  = (float*)   alloc((size_t)Bn * Hn * 4);
  _Float16* h1h  = (_Float16*)alloc((size_t)Bn * Hn * 2);
  _Float16* s1h  = (_Float16*)alloc((size_t)Bn * Hn * 2);
  _Float16* v2h  = (_Float16*)alloc((size_t)Bn * Hn * 2);
  float*    wf   = (float*)   alloc((size_t)Bn * Hn * 4);
  float*    Rf   = (float*)   alloc((size_t)Bn * Hn * 4);
  float*    lapA = (float*)   alloc((size_t)Bn * 4);

  k0_prep<<<(Hn * Hn + 255) / 256, 256, 0, stream>>>(W1, W2, w2h, w2th, w1h, q);
  k1_fwd<<<(Bn * Hn + 255) / 256, 256, 0, stream>>>(x, W1, b1, h1f, h1h, s1h, lapA);
  // 256 x 16 waves of 32x32 tiles, 4 waves/block
  k2_gemm1<<<(256 * 16) / 4, 128, 0, stream>>>(h1h, w2th, b2, W3, v2h, wf);
  k3_gemm2<<<(256 * 16) / 4, 128, 0, stream>>>(v2h, w2h, Rf);
  // 512 x 32 waves of 16x16 tiles, 4 waves/block
  k5_lap<<<(512 * 32) / 4, 128, 0, stream>>>(s1h, w2th, w1h, wf, lapA);
  k4_final<<<Bn / 8, 256, 0, stream>>>(x, W1, q, h1f, Rf, lapA, (float*)d_out);
}
